// MultiHeadAttention_29145648070761
// MI455X (gfx1250) — matrix operand order repaired and checked by reading, not compiled, not run on hardware
//
#include <hip/hip_runtime.h>
#include <hip/hip_bf16.h>

typedef __attribute__((ext_vector_type(16))) _Float16 v16h;
typedef __attribute__((ext_vector_type(8)))  _Float16 v8h;
typedef __attribute__((ext_vector_type(8)))  float    v8f;

#define EMB 1024
#define SEQ 2048
#define NBATCH 4
#define NHEAD 16
#define DHEAD 64
#define MROWS (NBATCH * SEQ)   // 8192

// ---------------------------------------------------------------------------
// A-fragment loader: 16x32 f16 A-matrix, lane l holds row (l&15).
// element e <-> k = e + (lane>=16 ? 16 : 0)
// => one contiguous 16-half (32B) chunk at kb+lh*16.
// ---------------------------------------------------------------------------
static __device__ __forceinline__ v16h load_a_frag(const _Float16* __restrict__ row,
                                                   int kb, int lh) {
  return *(const v16h*)(row + kb + lh * 16);
}

static __device__ __forceinline__ v8f wmma_f16(v16h a, v16h b, v8f c) {
  return __builtin_amdgcn_wmma_f32_16x16x32_f16(false, a, false, b, (short)0, c,
                                                false, false);
}

// ---------------------------------------------------------------------------
// Elementwise f32 -> f16 convert
// ---------------------------------------------------------------------------
__global__ __launch_bounds__(256) void cvt_f16_kernel(const float* __restrict__ in,
                                                      _Float16* __restrict__ out,
                                                      int n) {
  int i = blockIdx.x * 256 + threadIdx.x;
  if (i < n) out[i] = (_Float16)in[i];
}

// W [k][n] f32 -> WT [n][k] f16  (1024x1024)
__global__ __launch_bounds__(256) void cvt_wt_kernel(const float* __restrict__ W,
                                                     _Float16* __restrict__ WT) {
  int i = blockIdx.x * 256 + threadIdx.x;   // i = n*1024 + k
  int nn = i >> 10, kk = i & 1023;
  WT[i] = (_Float16)W[kk * EMB + nn];
}

// ---------------------------------------------------------------------------
// WMMA GEMM: out(+bias) = X[8192x1024] @ W (W given transposed: WT[n][k]).
// One wave computes a 16(M) x 64(N) tile: 32 K-steps x 4 wmma.
// MODE 0: f16 per-head row-major [nb][h][s][d]               (K)
// MODE 1: f16 per-head row-major, pre-scaled by 1/sqrt(EMB)  (Q)
// MODE 2: f16 per-head d-major   [nb][h][d][s]               (V)
// MODE 3: f32 row-major [8192][1024]                         (final output)
// ---------------------------------------------------------------------------
template <int MODE>
__global__ __launch_bounds__(256) void proj_gemm_kernel(
    const _Float16* __restrict__ X, const _Float16* __restrict__ WT,
    const float* __restrict__ bias, _Float16* __restrict__ outH,
    float* __restrict__ outF) {
  const int lane = threadIdx.x & 31;
  const int wave = blockIdx.x * 8 + (threadIdx.x >> 5);   // 8192 waves
  const int lh = lane >> 4, ll = lane & 15;
  const int m0 = (wave >> 4) << 4;   // 512 M-tiles
  const int n0 = (wave & 15) << 6;   // 16 N-supertiles of 64

  v8f acc[4] = {};
  const _Float16* arow = X + (size_t)(m0 + ll) * EMB;

  for (int kb = 0; kb < EMB; kb += 32) {
    v16h af = load_a_frag(arow, kb, lh);
#pragma unroll
    for (int j = 0; j < 4; ++j) {
      // B: lane l holds col (n0+j*16+ll); element e <-> k = kb + lh*16 + e
      const _Float16* bp = WT + (size_t)(n0 + j * 16 + ll) * EMB + kb + lh * 16;
      v16h bf = *(const v16h*)bp;
      acc[j] = wmma_f16(af, bf, acc[j]);
    }
  }

#pragma unroll
  for (int j = 0; j < 4; ++j) {
    const int n = n0 + j * 16 + ll;
    const float bv = bias[n];
#pragma unroll
    for (int r = 0; r < 8; ++r) {
      const int m = m0 + r + lh * 8;
      float v = acc[j][r] + bv;
      if (MODE == 1) v *= 0.03125f;   // fold 1/sqrt(EMB)=1/32 into Q (exact)
      if (MODE == 3) {
        outF[(size_t)m * EMB + n] = v;
      } else {
        const int nb = m >> 11, s = m & (SEQ - 1);
        const int h = n >> 6, d = n & 63;
        if (MODE == 2)
          outH[(((size_t)(nb * NHEAD + h) * DHEAD) + d) * SEQ + s] = (_Float16)v;
        else
          outH[(((size_t)(nb * NHEAD + h) * SEQ) + s) * DHEAD + d] = (_Float16)v;
      }
    }
  }
}

// ---------------------------------------------------------------------------
// Flash attention: one wave owns 16 query rows of one (nb, h).
// Loop over 32 key-tiles of 64: QK^T (8 wmma) -> online softmax (row
// reductions via xor-shuffle within 16-lane halves) -> P routed through a
// per-wave 16x64 LDS tile (C-layout -> A-layout) -> P@V (8 wmma) into
// 16x64 f32 accumulators. Q is pre-scaled by 1/sqrt(EMB) at projection time.
// ---------------------------------------------------------------------------
__global__ __launch_bounds__(256) void attn_kernel(
    const _Float16* __restrict__ Q,   // [nb][h][s][64]  (pre-scaled)
    const _Float16* __restrict__ K,   // [nb][h][s][64]
    const _Float16* __restrict__ Vt,  // [nb][h][64][s]
    _Float16* __restrict__ O) {       // [8192][1024] f16
  __shared__ _Float16 lds[8 * 1024];  // 16x64 f16 P-tile per wave (16KB)
  const int lane = threadIdx.x & 31;
  const int wib = threadIdx.x >> 5;
  const int wave = blockIdx.x * 8 + wib;   // 8192 waves = 4*16*128
  const int lh = lane >> 4, ll = lane & 15;

  const int nb = wave >> 11;
  const int h = (wave >> 7) & 15;
  const int mbase = (wave & 127) << 4;

  const _Float16* Qh = Q + (size_t)(nb * NHEAD + h) * SEQ * DHEAD;
  const _Float16* Kh = K + (size_t)(nb * NHEAD + h) * SEQ * DHEAD;
  const _Float16* Vh = Vt + (size_t)(nb * NHEAD + h) * DHEAD * SEQ;

  // Q fragments for this 16-row tile (K-dim 64 = two 32-chunks)
  const _Float16* qrow = Qh + (size_t)(mbase + ll) * DHEAD;
  const v16h qf0 = load_a_frag(qrow, 0, lh);
  const v16h qf1 = load_a_frag(qrow, 32, lh);

  v8f oacc[4] = {};
  float mrun[8], lrun[8];
#pragma unroll
  for (int r = 0; r < 8; ++r) { mrun[r] = -__builtin_inff(); lrun[r] = 0.0f; }

  _Float16* pt = lds + wib * 1024;

  for (int kb = 0; kb < SEQ; kb += 64) {
    // ---- scores: 4 column subtiles of 16 keys each --------------------
    v8f sc[4];
#pragma unroll
    for (int c = 0; c < 4; ++c) {
      // B for QK^T: B[k=d][n=key] = K[key][d]; contiguous along d.
      const _Float16* kc = Kh + (size_t)(kb + c * 16 + ll) * DHEAD;
      v16h b0 = *(const v16h*)(kc + lh * 16);
      v16h b1 = *(const v16h*)(kc + 32 + lh * 16);
      v8f s = {};
      s = wmma_f16(qf0, b0, s);
      s = wmma_f16(qf1, b1, s);
      sc[c] = s;
    }
    // Prefetch next iteration's K/V tiles into the cache hierarchy
    if (kb + 64 < SEQ) {
      __builtin_prefetch(Kh + (size_t)(kb + 64 + ll) * DHEAD, 0, 3);
      __builtin_prefetch(Vh + (size_t)ll * SEQ + kb + 64, 0, 3);
    }
    // ---- online softmax ----------------------------------------------
#pragma unroll
    for (int r = 0; r < 8; ++r) {
      const float a0 = sc[0][r], a1 = sc[1][r];
      const float a2 = sc[2][r], a3 = sc[3][r];
      float rv = fmaxf(fmaxf(a0, a1), fmaxf(a2, a3));
      rv = fmaxf(rv, __shfl_xor(rv, 1, 32));
      rv = fmaxf(rv, __shfl_xor(rv, 2, 32));
      rv = fmaxf(rv, __shfl_xor(rv, 4, 32));
      rv = fmaxf(rv, __shfl_xor(rv, 8, 32));
      const float mnew = fmaxf(mrun[r], rv);
      const float corr = __expf(mrun[r] - mnew);
      const float p0 = __expf(a0 - mnew);
      const float p1 = __expf(a1 - mnew);
      const float p2 = __expf(a2 - mnew);
      const float p3 = __expf(a3 - mnew);
      float ps = (p0 + p1) + (p2 + p3);
      ps += __shfl_xor(ps, 1, 32);
      ps += __shfl_xor(ps, 2, 32);
      ps += __shfl_xor(ps, 4, 32);
      ps += __shfl_xor(ps, 8, 32);
      lrun[r] = lrun[r] * corr + ps;
      mrun[r] = mnew;
#pragma unroll
      for (int j = 0; j < 4; ++j) oacc[j][r] *= corr;
      // stash P in row-major 16x64 LDS tile (C layout -> memory)
      const int m = r + lh * 8;
      pt[m * 64 + ll]      = (_Float16)p0;
      pt[m * 64 + 16 + ll] = (_Float16)p1;
      pt[m * 64 + 32 + ll] = (_Float16)p2;
      pt[m * 64 + 48 + ll] = (_Float16)p3;
    }
    // Per-wave DS ops are in-order; fence so the A-layout reload sees them.
    asm volatile("s_wait_dscnt 0x0" ::: "memory");
    // ---- reload P as two A-fragments (k=0..31, k=32..63) -------------
    const v16h pf0 = load_a_frag(pt + ll * 64, 0, lh);
    const v16h pf1 = load_a_frag(pt + ll * 64, 32, lh);
    // ---- P(16x64) @ V(64x64): B[k=key][n=d] = Vt[d][key] -------------
#pragma unroll
    for (int j = 0; j < 4; ++j) {
      const _Float16* vp = Vh + (size_t)(j * 16 + ll) * SEQ + kb;
      v16h vf0 = *(const v16h*)(vp + lh * 16);
      v16h vf1 = *(const v16h*)(vp + 32 + lh * 16);
      oacc[j] = wmma_f16(pf0, vf0, oacc[j]);
      oacc[j] = wmma_f16(pf1, vf1, oacc[j]);
    }
  }

  // Normalize and store to [8192][1024] f16 (input of the Wo projection)
#pragma unroll
  for (int r = 0; r < 8; ++r) {
    const float inv = 1.0f / lrun[r];
    const int s = mbase + r + lh * 8;
    const size_t orow = ((size_t)nb * SEQ + s) * EMB + h * DHEAD;
#pragma unroll
    for (int j = 0; j < 4; ++j) {
      O[orow + j * 16 + ll] = (_Float16)(oacc[j][r] * inv);
    }
  }
}

// ---------------------------------------------------------------------------
// Host-side launcher
// ---------------------------------------------------------------------------
extern "C" void kernel_launch(void* const* d_in, const int* in_sizes, int n_in,
                              void* d_out, int out_size, void* d_ws,
                              size_t ws_size, hipStream_t stream) {
  (void)in_sizes; (void)n_in; (void)out_size; (void)ws_size;

  const float* a  = (const float*)d_in[0];
  const float* Wq = (const float*)d_in[1];
  const float* bq = (const float*)d_in[2];
  const float* Wk = (const float*)d_in[3];
  const float* bk = (const float*)d_in[4];
  const float* Wv = (const float*)d_in[5];
  const float* bv = (const float*)d_in[6];
  const float* Wo = (const float*)d_in[7];
  const float* bo = (const float*)d_in[8];
  float* out = (float*)d_out;

  // Workspace carve-up (f16 elements): ~88 MiB total
  _Float16* ws  = (_Float16*)d_ws;
  const size_t XSZ = (size_t)MROWS * EMB;       // 8388608
  const size_t WSZ = (size_t)EMB * EMB;         // 1048576
  _Float16* X16 = ws;
  _Float16* WqT = X16 + XSZ;
  _Float16* WkT = WqT + WSZ;
  _Float16* WvT = WkT + WSZ;
  _Float16* WoT = WvT + WSZ;
  _Float16* Q16 = WoT + WSZ;
  _Float16* K16 = Q16 + XSZ;
  _Float16* V16 = K16 + XSZ;
  _Float16* X2  = V16 + XSZ;

  // 1) Converts
  cvt_f16_kernel<<<(int)(XSZ / 256), 256, 0, stream>>>(a, X16, (int)XSZ);
  cvt_wt_kernel<<<(int)(WSZ / 256), 256, 0, stream>>>(Wq, WqT);
  cvt_wt_kernel<<<(int)(WSZ / 256), 256, 0, stream>>>(Wk, WkT);
  cvt_wt_kernel<<<(int)(WSZ / 256), 256, 0, stream>>>(Wv, WvT);
  cvt_wt_kernel<<<(int)(WSZ / 256), 256, 0, stream>>>(Wo, WoT);

  // 2) Q/K/V projections (8192 waves -> 1024 blocks of 8 waves)
  proj_gemm_kernel<1><<<1024, 256, 0, stream>>>(X16, WqT, bq, Q16, nullptr);
  proj_gemm_kernel<0><<<1024, 256, 0, stream>>>(X16, WkT, bk, K16, nullptr);
  proj_gemm_kernel<2><<<1024, 256, 0, stream>>>(X16, WvT, bv, V16, nullptr);

  // 3) Flash attention (4*16*128 = 8192 waves)
  attn_kernel<<<1024, 256, 0, stream>>>(Q16, K16, V16, X2);

  // 4) Output projection (f32 result + bias)
  proj_gemm_kernel<3><<<1024, 256, 0, stream>>>(X2, WoT, bo, nullptr, out);
}
